// SparseAttention_29171417875242
// MI455X (gfx1250) — matrix operand order repaired and checked by reading, not compiled, not run on hardware
//
#include <hip/hip_runtime.h>
#include <hip/hip_bf16.h>
#include <math.h>

#define BATCH 4
#define SEQ   2048
#define NHEAD 16
#define HD    64
#define DM    1024
#define MR    (BATCH*SEQ)      // 8192 rows
#define KSEL  614              // int(2048*0.3)
#define KPAD  640              // 20 tiles of 32

typedef __bf16 bf16;
typedef __attribute__((ext_vector_type(16))) bf16  v16bf;
typedef __attribute__((ext_vector_type(8)))  bf16  v8bf;
typedef __attribute__((ext_vector_type(8)))  float v8f;

union V16U { v16bf v; v8bf h[2]; };

static __device__ __forceinline__ v16bf load2x8(const bf16* p0, const bf16* p1) {
  V16U u;
  u.h[0] = *(const v8bf*)p0;
  u.h[1] = *(const v8bf*)p1;
  return u.v;
}

static __device__ __forceinline__ v8f wmma_bf16(v16bf a, v16bf b, v8f c) {
  // D = A(16x32 bf16) * B(32x16 bf16) + C(16x16 f32)
  return __builtin_amdgcn_wmma_f32_16x16x32_bf16(false, a, false, b, (short)0, c,
                                                 false, false);
}

// ---------------------------------------------------------------------------
// fp32 -> bf16 elementwise convert
// ---------------------------------------------------------------------------
__global__ void cvt_f32_bf16(const float* __restrict__ in, bf16* __restrict__ out, int n) {
  int i = blockIdx.x * blockDim.x + threadIdx.x;
  if (i < n) out[i] = (bf16)in[i];
}

// ---------------------------------------------------------------------------
// C[M,N] = A[M,K] * W[N,K]^T + bias   (torch Linear), bf16 inputs, WMMA core.
// WG: 256 threads = 8 waves, arranged 2(M) x 4(N); wave tile 32x32 (2x2 frags).
// WG tile = 64(M) x 128(N).
// ---------------------------------------------------------------------------
template<bool OUT_F32>
__global__ __launch_bounds__(256)
void gemm_xwt(const bf16* __restrict__ A, const bf16* __restrict__ W,
              const float* __restrict__ bias, void* __restrict__ Cout,
              int M, int N, int Kd) {
  const int lane = threadIdx.x & 31;
  const int wave = threadIdx.x >> 5;
  const int ln   = lane & 15;
  const int hl   = lane >> 4;          // half-wave: 0 or 1

  const int mtile = blockIdx.y * 64  + (wave >> 2) * 32;
  const int ntile = blockIdx.x * 128 + (wave & 3)  * 32;

  const bf16* Arow0 = A + (size_t)(mtile      + ln) * Kd;
  const bf16* Arow1 = A + (size_t)(mtile + 16 + ln) * Kd;
  const bf16* Wrow0 = W + (size_t)(ntile      + ln) * Kd;
  const bf16* Wrow1 = W + (size_t)(ntile + 16 + ln) * Kd;

  const int aoff = hl * 8;    // A frag: chunks at k+aoff and k+aoff+16
  const int boff = hl * 8;    // B frag: chunks at k+boff and k+boff+16

  v8f acc[2][2] = {};

  for (int kk = 0; kk < Kd; kk += 32) {
    if (kk + 32 < Kd) {                      // prefetch next K tile
      __builtin_prefetch(Arow0 + kk + 32, 0, 3);
      __builtin_prefetch(Wrow0 + kk + 32, 0, 3);
    }
    v16bf a0 = load2x8(Arow0 + kk + aoff, Arow0 + kk + aoff + 16);
    v16bf a1 = load2x8(Arow1 + kk + aoff, Arow1 + kk + aoff + 16);
    v16bf b0 = load2x8(Wrow0 + kk + boff, Wrow0 + kk + boff + 16);
    v16bf b1 = load2x8(Wrow1 + kk + boff, Wrow1 + kk + boff + 16);
    acc[0][0] = wmma_bf16(a0, b0, acc[0][0]);
    acc[0][1] = wmma_bf16(a0, b1, acc[0][1]);
    acc[1][0] = wmma_bf16(a1, b0, acc[1][0]);
    acc[1][1] = wmma_bf16(a1, b1, acc[1][1]);
  }

  #pragma unroll
  for (int i = 0; i < 2; i++) {
    #pragma unroll
    for (int j = 0; j < 2; j++) {
      const int n = ntile + j * 16 + ln;
      const float bv = bias ? bias[n] : 0.0f;
      #pragma unroll
      for (int r = 0; r < 8; r++) {
        const int m = mtile + i * 16 + r + hl * 8;   // C layout: row = r + 8*hl
        float v = acc[i][j][r] + bv;
        if (OUT_F32) ((float*)Cout)[(size_t)m * N + n] = v;
        else         ((bf16*)Cout)[(size_t)m * N + n] = (bf16)v;
      }
    }
  }
}

// ---------------------------------------------------------------------------
// Indexer (kept fp32 so top-k selection matches the reference):
// imp[row] = sum_j relu(x_row . wi1_j + bi1_j) * wi2_j + bi2
// ---------------------------------------------------------------------------
__global__ __launch_bounds__(256)
void indexer_kernel(const float* __restrict__ x, const float* __restrict__ wi1,
                    const float* __restrict__ bi1, const float* __restrict__ wi2,
                    const float* __restrict__ bi2, float* __restrict__ imp) {
  __shared__ float xs[DM];
  __shared__ float red[256];
  const int row = blockIdx.x;
  const float* xr = x + (size_t)row * DM;
  for (int i = threadIdx.x; i < DM; i += 256) xs[i] = xr[i];
  __syncthreads();

  float acc = 0.0f;
  for (int j = threadIdx.x; j < DM / 2; j += 256) {
    const float* wr = wi1 + (size_t)j * DM;
    float d = bi1[j];
    #pragma unroll 4
    for (int t = 0; t < DM; t += 4) {
      d += xs[t] * wr[t] + xs[t+1] * wr[t+1] + xs[t+2] * wr[t+2] + xs[t+3] * wr[t+3];
    }
    acc += fmaxf(d, 0.0f) * wi2[j];
  }
  red[threadIdx.x] = acc;
  __syncthreads();
  for (int s = 128; s > 0; s >>= 1) {
    if (threadIdx.x < (unsigned)s) red[threadIdx.x] += red[threadIdx.x + s];
    __syncthreads();
  }
  if (threadIdx.x == 0) imp[row] = red[0] + bi2[0];
}

// ---------------------------------------------------------------------------
// Top-k by ranking (count-greater with index tie-break == jax top_k set).
// Scatters selected positions into a compact, rank-ordered index list.
// ---------------------------------------------------------------------------
__global__ __launch_bounds__(256)
void topk_kernel(const float* __restrict__ imp, int* __restrict__ idx) {
  __shared__ float si[SEQ];
  const int b = blockIdx.x;
  const float* ib = imp + (size_t)b * SEQ;
  for (int i = threadIdx.x; i < SEQ; i += 256) si[i] = ib[i];
  __syncthreads();
  for (int s = threadIdx.x; s < SEQ; s += 256) {
    const float v = si[s];
    int cnt = 0;
    for (int t = 0; t < SEQ; t++) {
      const float u = si[t];
      cnt += (u > v) || (u == v && t < s);
    }
    if (cnt < KSEL) idx[b * KPAD + cnt] = s;
  }
}

// ---------------------------------------------------------------------------
// Gather selected keys/values per (b,h):
//   Ksel[bh][j][d] = K[b, idx[b][j], h*64+d]       (row per key, contiguous hd)
//   Vt  [bh][d][j] = V[b, idx[b][j], h*64+d]       (transposed, contiguous keys)
// Padding keys (j >= KSEL) are zero-filled.
// ---------------------------------------------------------------------------
__global__ __launch_bounds__(256)
void gather_kernel(const bf16* __restrict__ Kp, const bf16* __restrict__ Vp,
                   const int* __restrict__ idx, bf16* __restrict__ Ks,
                   bf16* __restrict__ Vt) {
  const int bh = blockIdx.x;
  const int b = bh >> 4, h = bh & 15;
  const int* ib = idx + b * KPAD;
  bf16* Kd = Ks + (size_t)bh * KPAD * HD;
  bf16* Vd = Vt + (size_t)bh * HD * KPAD;
  for (int e = threadIdx.x; e < KPAD * HD; e += 256) {
    const int j = e / HD, d = e % HD;
    bf16 kv = (bf16)0.0f, vv = (bf16)0.0f;
    if (j < KSEL) {
      const size_t src = ((size_t)(b * SEQ + ib[j])) * DM + h * HD + d;
      kv = Kp[src];
      vv = Vp[src];
    }
    Kd[(size_t)j * HD + d]   = kv;
    Vd[(size_t)d * KPAD + j] = vv;
  }
}

// ---------------------------------------------------------------------------
// Flash-style sparse attention. One wave = 16 queries of one (b,h).
// Per 32-key tile: 4 WMMA for Q*K^T, online softmax (fp32), P transposed via
// per-wave LDS tile into A-fragment layout, 4 WMMA for P*V.
// ---------------------------------------------------------------------------
__global__ __launch_bounds__(256)
void attn_kernel(const bf16* __restrict__ Qb, const bf16* __restrict__ Ksel,
                 const bf16* __restrict__ Vt, bf16* __restrict__ Ao) {
  __shared__ __align__(16) bf16 pt[8][16 * 32];   // P tile per wave (1 KB each)

  const int lane = threadIdx.x & 31;
  const int wave = threadIdx.x >> 5;
  const int ln   = lane & 15;
  const int hl   = lane >> 4;

  const int bh    = blockIdx.x >> 4;            // (b,h) pair: 0..63
  const int qblk  = blockIdx.x & 15;            // 16 blocks per (b,h)
  const int b     = bh >> 4, h = bh & 15;
  const int qbase = qblk * 128 + wave * 16;
  const float scale = 0.125f;                   // 1/sqrt(64)

  // Q A-fragments for the two hd K-steps (k = 0..31, 32..63)
  const bf16* Qrow = Qb + ((size_t)(b * SEQ + qbase + ln)) * DM + h * HD;
  v16bf aq[2];
  #pragma unroll
  for (int s = 0; s < 2; s++) {
    const int k0 = s * 32 + hl * 8;
    aq[s] = load2x8(Qrow + k0, Qrow + k0 + 16);
  }

  const bf16* Kb_ = Ksel + (size_t)bh * KPAD * HD;
  const bf16* Vb_ = Vt   + (size_t)bh * HD * KPAD;
  bf16* lp = pt[wave];

  v8f o[4] = {};
  float mrow[8], lrow[8];
  #pragma unroll
  for (int r = 0; r < 8; r++) { mrow[r] = -__builtin_inff(); lrow[r] = 0.0f; }

  for (int kt = 0; kt < KPAD; kt += 32) {
    // ---- scores: S(16x32) = Q(16x64) * Ksel_tile^T ----
    v8f sf[2] = {};
    #pragma unroll
    for (int sub = 0; sub < 2; sub++) {
      const bf16* Krow = Kb_ + (size_t)(kt + sub * 16 + ln) * HD + hl * 8;
      #pragma unroll
      for (int s = 0; s < 2; s++) {
        v16bf kb = load2x8(Krow + s * 32, Krow + s * 32 + 16);
        sf[sub] = wmma_bf16(aq[s], kb, sf[sub]);
      }
    }
    // ---- scale + mask padded / unselected columns ----
    #pragma unroll
    for (int sub = 0; sub < 2; sub++) {
      const int key = kt + sub * 16 + ln;
      const float msk = (key < KSEL) ? 0.0f : -__builtin_inff();
      #pragma unroll
      for (int r = 0; r < 8; r++) sf[sub][r] = sf[sub][r] * scale + msk;
    }
    // ---- row max (reduce across the 16 lanes of each half-wave) ----
    float t[8];
    #pragma unroll
    for (int r = 0; r < 8; r++) t[r] = fmaxf(sf[0][r], sf[1][r]);
    #pragma unroll
    for (int off = 1; off < 16; off <<= 1)
      #pragma unroll
      for (int r = 0; r < 8; r++) t[r] = fmaxf(t[r], __shfl_xor(t[r], off, 32));

    float alpha[8];
    #pragma unroll
    for (int r = 0; r < 8; r++) {
      const float mn = fmaxf(mrow[r], t[r]);
      alpha[r] = __expf(mrow[r] - mn);
      mrow[r]  = mn;
      lrow[r] *= alpha[r];
    }
    #pragma unroll
    for (int j = 0; j < 4; j++)
      #pragma unroll
      for (int r = 0; r < 8; r++) o[j][r] *= alpha[r];

    // ---- P = exp(S - m) ----
    #pragma unroll
    for (int sub = 0; sub < 2; sub++)
      #pragma unroll
      for (int r = 0; r < 8; r++) sf[sub][r] = __expf(sf[sub][r] - mrow[r]);

    // ---- row sum ----
    #pragma unroll
    for (int r = 0; r < 8; r++) t[r] = sf[0][r] + sf[1][r];
    #pragma unroll
    for (int off = 1; off < 16; off <<= 1)
      #pragma unroll
      for (int r = 0; r < 8; r++) t[r] += __shfl_xor(t[r], off, 32);
    #pragma unroll
    for (int r = 0; r < 8; r++) lrow[r] += t[r];

    // ---- transpose P (C layout -> A layout) through per-wave LDS tile ----
    #pragma unroll
    for (int sub = 0; sub < 2; sub++)
      #pragma unroll
      for (int r = 0; r < 8; r++)
        lp[(r + hl * 8) * 32 + sub * 16 + ln] = (bf16)sf[sub][r];
    // LDS ops are in-order within a wave; compiler inserts s_wait_dscnt.
    const int pk0 = hl * 8;
    v16bf ap = load2x8(lp + ln * 32 + pk0, lp + ln * 32 + pk0 + 16);

    // ---- O += P(16x32) * V_tile(32x64) ----
    #pragma unroll
    for (int j = 0; j < 4; j++) {
      const bf16* Vrow = Vb_ + (size_t)(j * 16 + ln) * KPAD + kt + hl * 8;
      v16bf vb = load2x8(Vrow, Vrow + 16);
      o[j] = wmma_bf16(ap, vb, o[j]);
    }
  }

  // ---- normalize and write (bf16, natural [B,S,D] layout) ----
  #pragma unroll
  for (int r = 0; r < 8; r++) {
    const float inv = 1.0f / lrow[r];
    const size_t rowoff = ((size_t)(b * SEQ + qbase + r + hl * 8)) * DM + h * HD;
    #pragma unroll
    for (int j = 0; j < 4; j++)
      Ao[rowoff + j * 16 + ln] = (bf16)(o[j][r] * inv);
  }
}

// ---------------------------------------------------------------------------
// Host-side launcher
// ---------------------------------------------------------------------------
extern "C" void kernel_launch(void* const* d_in, const int* in_sizes, int n_in,
                              void* d_out, int out_size, void* d_ws, size_t ws_size,
                              hipStream_t stream) {
  (void)in_sizes; (void)n_in; (void)out_size; (void)ws_size;
  const float* x   = (const float*)d_in[0];
  const float* wq  = (const float*)d_in[1];
  const float* bq  = (const float*)d_in[2];
  const float* wk  = (const float*)d_in[3];
  const float* bk  = (const float*)d_in[4];
  const float* wv  = (const float*)d_in[5];
  const float* bv  = (const float*)d_in[6];
  const float* wo  = (const float*)d_in[7];
  const float* bo  = (const float*)d_in[8];
  const float* wi1 = (const float*)d_in[9];
  const float* bi1 = (const float*)d_in[10];
  const float* wi2 = (const float*)d_in[11];
  const float* bi2 = (const float*)d_in[12];
  float* out = (float*)d_out;

  // workspace carve-out (256B aligned slices)
  char* ws = (char*)d_ws;
  size_t off = 0;
  auto alloc = [&](size_t bytes) -> char* {
    char* p = ws + off;
    off += (bytes + 255) & ~(size_t)255;
    return p;
  };
  bf16* xb  = (bf16*)alloc((size_t)MR * DM * 2);
  bf16* wqb = (bf16*)alloc((size_t)DM * DM * 2);
  bf16* wkb = (bf16*)alloc((size_t)DM * DM * 2);
  bf16* wvb = (bf16*)alloc((size_t)DM * DM * 2);
  bf16* wob = (bf16*)alloc((size_t)DM * DM * 2);
  bf16* Qb  = (bf16*)alloc((size_t)MR * DM * 2);
  bf16* Kb  = (bf16*)alloc((size_t)MR * DM * 2);
  bf16* Vb  = (bf16*)alloc((size_t)MR * DM * 2);
  bf16* Ab  = (bf16*)alloc((size_t)MR * DM * 2);
  float* imp = (float*)alloc((size_t)MR * 4);
  int*   idx = (int*)alloc((size_t)BATCH * KPAD * 4);
  bf16* Ks  = (bf16*)alloc((size_t)BATCH * NHEAD * KPAD * HD * 2);
  bf16* Vt  = (bf16*)alloc((size_t)BATCH * NHEAD * HD * KPAD * 2);

  // 1) fp32 -> bf16 converts
  {
    int n = MR * DM;
    cvt_f32_bf16<<<(n + 255) / 256, 256, 0, stream>>>(x, xb, n);
    n = DM * DM;
    cvt_f32_bf16<<<(n + 255) / 256, 256, 0, stream>>>(wq, wqb, n);
    cvt_f32_bf16<<<(n + 255) / 256, 256, 0, stream>>>(wk, wkb, n);
    cvt_f32_bf16<<<(n + 255) / 256, 256, 0, stream>>>(wv, wvb, n);
    cvt_f32_bf16<<<(n + 255) / 256, 256, 0, stream>>>(wo, wob, n);
  }

  // 2) Q/K/V projections (WMMA, bf16 out)
  dim3 ggrid(DM / 128, MR / 64);
  gemm_xwt<false><<<ggrid, 256, 0, stream>>>(xb, wqb, bq, Qb, MR, DM, DM);
  gemm_xwt<false><<<ggrid, 256, 0, stream>>>(xb, wkb, bk, Kb, MR, DM, DM);
  gemm_xwt<false><<<ggrid, 256, 0, stream>>>(xb, wvb, bv, Vb, MR, DM, DM);

  // 3) fp32 indexer -> importance
  indexer_kernel<<<MR, 256, 0, stream>>>(x, wi1, bi1, wi2, bi2, imp);

  // 4) top-k selection per batch
  topk_kernel<<<BATCH, 256, 0, stream>>>(imp, idx);

  // 5) gather selected K / transposed V
  gather_kernel<<<BATCH * NHEAD, 256, 0, stream>>>(Kb, Vb, idx, Ks, Vt);

  // 6) sparse flash attention (WMMA)
  attn_kernel<<<BATCH * NHEAD * 16, 256, 0, stream>>>(Qb, Ks, Vt, Ab);

  // 7) output projection (WMMA, fp32 out)
  gemm_xwt<true><<<ggrid, 256, 0, stream>>>(Ab, wob, bo, out, MR, DM, DM);
}
